// SocialModel_89756226551851
// MI455X (gfx1250) — matrix operand order repaired and checked by reading, not compiled, not run on hardware
//
#include <hip/hip_runtime.h>
#include <cstddef>

// ---------------------------------------------------------------------------
// Social-LSTM step for MI455X (gfx1250, wave32, WMMA bf16).
// T=19, N=512, GG=16, R=128, E=64, IN=2, OUT=5
// ---------------------------------------------------------------------------

#define T_STEPS 19
#define N_NODES 512
#define G_CELLS 16
#define R_DIM   128
#define E_DIM   64

typedef __bf16 bf16_t;
typedef __attribute__((ext_vector_type(16))) __bf16 v16bf;
typedef __attribute__((ext_vector_type(8)))  __bf16 v8bf;
typedef __attribute__((ext_vector_type(8)))  float  v8f;
typedef __attribute__((ext_vector_type(4)))  float  v4f;

static __device__ __forceinline__ v16bf join16(v8bf lo, v8bf hi) {
  v16bf r;
#pragma unroll
  for (int i = 0; i < 8; ++i) { r[i] = lo[i]; r[8 + i] = hi[i]; }
  return r;
}

// A-fragment (16x32 bf16): lanes 0-15 hold K=0..15, lanes 16-31 K=16..31
// (contiguous 16 elements per lane); caller passes base at first K.
static __device__ __forceinline__ v16bf lds_loadA(const bf16_t* p) {
  return join16(*(const v8bf*)p, *(const v8bf*)(p + 8));
}

// B-fragment (32x16 bf16): lanes 0-15 hold K=0..15, lanes 16-31 K=16..31
// (contiguous 16 elements per lane).
static __device__ __forceinline__ v16bf lds_loadB(const bf16_t* p) {
  return join16(*(const v8bf*)p, *(const v8bf*)(p + 8));
}

// Load 16 contiguous f32 from global (L2-resident weights) and convert.
static __device__ __forceinline__ v16bf load_cvt16(const float* p) {
  v4f q0 = *(const v4f*)(p + 0);
  v4f q1 = *(const v4f*)(p + 4);
  v4f q2 = *(const v4f*)(p + 8);
  v4f q3 = *(const v4f*)(p + 12);
  v16bf r;
#pragma unroll
  for (int i = 0; i < 4; ++i) {
    r[i]      = (__bf16)q0[i];
    r[4 + i]  = (__bf16)q1[i];
    r[8 + i]  = (__bf16)q2[i];
    r[12 + i] = (__bf16)q3[i];
  }
  return r;
}

static __device__ __forceinline__ float sigf(float x) {
  return 1.0f / (1.0f + __expf(-x));
}

// Dynamic LDS layout (88 KB total):
//   region A: [0, 24KB)
//     phase 1 : s_grid bf16[16][16][32] (16KB) + s_hT bf16[128][32] (8KB)
//     later   : s_te f32[16][64] @0 (4KB) | s_xcat bf16[16][128] @4KB
//               | s_hblk bf16[16][128] @8KB | s_hnew f32[16][128] @12KB
//   region B: [24KB, 88KB)
//     phase 1-2: s_soc bf16[16][16][128] (64KB)
//     phase 4-5: s_gates f32[16][512] (32KB)
extern __shared__ char smem[];

__global__ __launch_bounds__(512, 1)
void social_lstm_step(int t,
                      const float* __restrict__ grids,   // [T][N][N][GG]
                      const float* __restrict__ xin,     // [T][N][2]
                      const float* __restrict__ h_cur,   // [N][R]
                      const float* __restrict__ c_cur,   // [N][R]
                      float* __restrict__ h_nxt,         // [N][R]
                      float* __restrict__ c_nxt,         // [N][R]
                      const float* __restrict__ Wi,      // [64][2]
                      const float* __restrict__ bi,      // [64]
                      const float* __restrict__ Wt,      // [64][2048]
                      const float* __restrict__ bt,      // [64]
                      const float* __restrict__ Wih,     // [512][128]
                      const float* __restrict__ Whh,     // [512][128]
                      const float* __restrict__ bih,     // [512]
                      const float* __restrict__ bhh,     // [512]
                      const float* __restrict__ Wo,      // [5][128]
                      const float* __restrict__ bo,      // [5]
                      float* __restrict__ out)           // [T][N][5]
{
  char* sA = smem;
  char* sB = smem + 24 * 1024;

  bf16_t* s_grid  = (bf16_t*)sA;               // [g][n][m]
  bf16_t* s_hT    = (bf16_t*)(sA + 16 * 1024); // [r][m]
  float*  s_te    = (float*)sA;                // [n][e]
  bf16_t* s_xcat  = (bf16_t*)(sA + 4 * 1024);  // [n][0..127] = ie|te
  bf16_t* s_hblk  = (bf16_t*)(sA + 8 * 1024);  // [n][r]
  float*  s_hnew  = (float*)(sA + 12 * 1024);  // [n][r]
  bf16_t* s_soc   = (bf16_t*)sB;               // [g][n][r]
  float*  s_gates = (float*)sB;                // [n][j]

  const int tid    = threadIdx.x;
  const int lane   = tid & 31;
  const int w      = tid >> 5;    // wave id == grid cell g
  const int ncol   = lane & 15;   // row (A/C) or column (B/C) index
  const int hiHalf = lane >> 4;
  const int n0     = blockIdx.x * 16;

  const int kA = hiHalf ? 16 : 0;  // A-fragment first-K offset
  const int kB = hiHalf ? 16 : 0;  // B-fragment first-K offset

  const float* gbase = grids + (size_t)t * N_NODES * N_NODES * G_CELLS;

  // ------------------------------------------------------------------
  // Phase 1: social pooling. acc[rt] = grid_g(16x512) @ h(512x128) tile
  // ------------------------------------------------------------------
  v8f acc[8];
#pragma unroll
  for (int i = 0; i < 8; ++i)
#pragma unroll
    for (int v = 0; v < 8; ++v) acc[i][v] = 0.0f;

  const int sn = tid >> 5;          // staging: node 0..15
  const int sm = tid & 31;          // staging: m     0..31
  const int hm = tid >> 4;          // h staging: m   0..31
  const int hr = (tid & 15) * 8;    // h staging: r base

  for (int mc = 0; mc < 16; ++mc) {
    const int m0 = mc * 32;
    // ---- cooperative staging (fp32 HBM read -> bf16 LDS) ----
    {
      const float* gp =
          gbase + ((size_t)(n0 + sn) * N_NODES + (size_t)(m0 + sm)) * G_CELLS;
      v4f gq0 = *(const v4f*)(gp + 0);
      v4f gq1 = *(const v4f*)(gp + 4);
      v4f gq2 = *(const v4f*)(gp + 8);
      v4f gq3 = *(const v4f*)(gp + 12);
#pragma unroll
      for (int i = 0; i < 4; ++i) {
        s_grid[(((0 + i) * 16 + sn) * 32) + sm]  = (__bf16)gq0[i];
        s_grid[(((4 + i) * 16 + sn) * 32) + sm]  = (__bf16)gq1[i];
        s_grid[(((8 + i) * 16 + sn) * 32) + sm]  = (__bf16)gq2[i];
        s_grid[(((12 + i) * 16 + sn) * 32) + sm] = (__bf16)gq3[i];
      }
      if (mc < 15) __builtin_prefetch(gp + 32 * G_CELLS, 0, 1); // next chunk
    }
    {
      const float* hp = h_cur + (size_t)(m0 + hm) * R_DIM + hr;
      v4f h0 = *(const v4f*)hp;
      v4f h1 = *(const v4f*)(hp + 4);
#pragma unroll
      for (int i = 0; i < 4; ++i) {
        s_hT[(hr + i) * 32 + hm]     = (__bf16)h0[i];
        s_hT[(hr + 4 + i) * 32 + hm] = (__bf16)h1[i];
      }
    }
    __syncthreads();

    // ---- WMMA: wave w consumes its own g-plane ----
    {
      v16bf a = lds_loadA(s_grid + ((w * 16 + ncol) * 32) + kA);
#pragma unroll
      for (int rt = 0; rt < 8; ++rt) {
        v16bf b = lds_loadB(s_hT + (rt * 16 + ncol) * 32 + kB);
        acc[rt] = __builtin_amdgcn_wmma_f32_16x16x32_bf16(
            false, a, false, b, (short)0, acc[rt], false, false);
      }
    }
    __syncthreads();
  }

  // ------------------------------------------------------------------
  // Write social (bf16) to LDS; init te accumulator with bias.
  // ------------------------------------------------------------------
#pragma unroll
  for (int rt = 0; rt < 8; ++rt)
#pragma unroll
    for (int v = 0; v < 8; ++v)
      s_soc[(w * 16 + v + hiHalf * 8) * R_DIM + rt * 16 + ncol] =
          (__bf16)acc[rt][v];
  {
    const int i0 = tid * 2; // 1024 entries [n][e]
    s_te[i0]     = bt[i0 & 63];
    s_te[i0 + 1] = bt[(i0 + 1) & 63];
  }
  __syncthreads();

  // ------------------------------------------------------------------
  // Phase 2: te partial per wave (K-split by g), ds_add_f32 reduction.
  // ------------------------------------------------------------------
  {
    v8f p[4];
#pragma unroll
    for (int et = 0; et < 4; ++et)
#pragma unroll
      for (int v = 0; v < 8; ++v) p[et][v] = 0.0f;

#pragma unroll
    for (int kt = 0; kt < 4; ++kt) {
      v16bf a = lds_loadA(s_soc + (w * 16 + ncol) * R_DIM + kt * 32 + kA);
#pragma unroll
      for (int et = 0; et < 4; ++et) {
        const float* wp = Wt + (size_t)(et * 16 + ncol) * (G_CELLS * R_DIM) +
                          w * R_DIM + kt * 32 + kB;
        v16bf b = load_cvt16(wp);
        p[et] = __builtin_amdgcn_wmma_f32_16x16x32_bf16(
            false, a, false, b, (short)0, p[et], false, false);
      }
    }
#pragma unroll
    for (int et = 0; et < 4; ++et)
#pragma unroll
      for (int v = 0; v < 8; ++v)
        atomicAdd(&s_te[(v + hiHalf * 8) * E_DIM + et * 16 + ncol], p[et][v]);
  }
  __syncthreads();

  // ------------------------------------------------------------------
  // Phase 3: ie (tiny), relu(te), build xcat + h-block as bf16.
  // ------------------------------------------------------------------
#pragma unroll
  for (int q = 0; q < 2; ++q) {
    const int idx = tid * 2 + q;     // 0..1023
    const int n = idx >> 6, e = idx & 63;
    const float x0 = xin[((size_t)t * N_NODES + n0 + n) * 2 + 0];
    const float x1 = xin[((size_t)t * N_NODES + n0 + n) * 2 + 1];
    const float ie = x0 * Wi[e * 2 + 0] + x1 * Wi[e * 2 + 1] + bi[e];
    s_xcat[n * 128 + e]      = (__bf16)fmaxf(ie, 0.0f);
    s_xcat[n * 128 + 64 + e] = (__bf16)fmaxf(s_te[n * E_DIM + e], 0.0f);
  }
#pragma unroll
  for (int q = 0; q < 4; ++q) {
    const int idx = tid + q * 512;   // 0..2047
    const int n = idx >> 7, r = idx & 127;
    s_hblk[n * 128 + r] = (__bf16)h_cur[(size_t)(n0 + n) * R_DIM + r];
  }
  __syncthreads();

  // ------------------------------------------------------------------
  // Phase 4: gates = xcat @ Wih^T + h @ Whh^T + (bih+bhh). 2 tiles/wave.
  // ------------------------------------------------------------------
#pragma unroll
  for (int jt = 0; jt < 2; ++jt) {
    const int j0 = (w * 2 + jt) * 16;
    const float bias = bih[j0 + ncol] + bhh[j0 + ncol];
    v8f g;
#pragma unroll
    for (int v = 0; v < 8; ++v) g[v] = bias;

#pragma unroll
    for (int kt = 0; kt < 4; ++kt) {
      v16bf a = lds_loadA(s_xcat + ncol * 128 + kt * 32 + kA);
      v16bf b = load_cvt16(Wih + (size_t)(j0 + ncol) * 128 + kt * 32 + kB);
      g = __builtin_amdgcn_wmma_f32_16x16x32_bf16(false, a, false, b, (short)0,
                                                  g, false, false);
    }
#pragma unroll
    for (int kt = 0; kt < 4; ++kt) {
      v16bf a = lds_loadA(s_hblk + ncol * 128 + kt * 32 + kA);
      v16bf b = load_cvt16(Whh + (size_t)(j0 + ncol) * 128 + kt * 32 + kB);
      g = __builtin_amdgcn_wmma_f32_16x16x32_bf16(false, a, false, b, (short)0,
                                                  g, false, false);
    }
#pragma unroll
    for (int v = 0; v < 8; ++v)
      s_gates[(v + hiHalf * 8) * 512 + j0 + ncol] = g[v];
  }
  __syncthreads();

  // ------------------------------------------------------------------
  // Phase 5: LSTM elementwise (i,f,g,o), update c/h.
  // ------------------------------------------------------------------
#pragma unroll
  for (int q = 0; q < 4; ++q) {
    const int idx = tid + q * 512;   // 0..2047
    const int n = idx >> 7, r = idx & 127;
    const float gi = s_gates[n * 512 + r];
    const float gf = s_gates[n * 512 + 128 + r];
    const float gg = s_gates[n * 512 + 256 + r];
    const float go = s_gates[n * 512 + 384 + r];
    const float co = c_cur[(size_t)(n0 + n) * R_DIM + r];
    const float cn = sigf(gf) * co + sigf(gi) * tanhf(gg);
    const float hn = sigf(go) * tanhf(cn);
    c_nxt[(size_t)(n0 + n) * R_DIM + r] = cn;
    h_nxt[(size_t)(n0 + n) * R_DIM + r] = hn;
    s_hnew[n * 128 + r] = hn;
  }
  __syncthreads();

  // ------------------------------------------------------------------
  // Phase 6: output layer (16 nodes x 5 outputs).
  // ------------------------------------------------------------------
  if (tid < 80) {
    const int n = tid / 5, o = tid % 5;
    float s = bo[o];
#pragma unroll 8
    for (int r = 0; r < R_DIM; ++r) s += s_hnew[n * 128 + r] * Wo[o * 128 + r];
    out[((size_t)t * N_NODES + n0 + n) * 5 + o] = s;
  }
}

extern "C" void kernel_launch(void* const* d_in, const int* in_sizes, int n_in,
                              void* d_out, int out_size, void* d_ws,
                              size_t ws_size, hipStream_t stream) {
  (void)in_sizes; (void)n_in; (void)out_size; (void)ws_size;
  const float* input_data = (const float*)d_in[0];
  const float* grids      = (const float*)d_in[1];
  const float* h_init     = (const float*)d_in[2];
  const float* c_init     = (const float*)d_in[3];
  const float* Wi   = (const float*)d_in[4];
  const float* bi   = (const float*)d_in[5];
  const float* Wt   = (const float*)d_in[6];
  const float* bt   = (const float*)d_in[7];
  const float* Wih  = (const float*)d_in[8];
  const float* Whh  = (const float*)d_in[9];
  const float* bih  = (const float*)d_in[10];
  const float* bhh  = (const float*)d_in[11];
  const float* Wo   = (const float*)d_in[12];
  const float* bo   = (const float*)d_in[13];
  float* out = (float*)d_out;

  const size_t HC = (size_t)N_NODES * R_DIM;  // 65536 floats
  float* ws = (float*)d_ws;
  float* hbuf[2] = {ws, ws + HC};
  float* cbuf[2] = {ws + 2 * HC, ws + 3 * HC};

  hipMemcpyAsync(hbuf[0], h_init, HC * sizeof(float),
                 hipMemcpyDeviceToDevice, stream);
  hipMemcpyAsync(cbuf[0], c_init, HC * sizeof(float),
                 hipMemcpyDeviceToDevice, stream);

  const size_t shmem = 88 * 1024;
  for (int t = 0; t < T_STEPS; ++t) {
    const int cur = t & 1, nxt = cur ^ 1;
    social_lstm_step<<<dim3(N_NODES / 16), dim3(512), shmem, stream>>>(
        t, grids, input_data, hbuf[cur], cbuf[cur], hbuf[nxt], cbuf[nxt],
        Wi, bi, Wt, bt, Wih, Whh, bih, bhh, Wo, bo, out);
  }

  // outputs[T][N][5] | h_fin[N][R] | c_fin[N][R], concatenated flat
  float* h_out = out + (size_t)T_STEPS * N_NODES * 5;
  float* c_out = h_out + HC;
  hipMemcpyAsync(h_out, hbuf[T_STEPS & 1], HC * sizeof(float),
                 hipMemcpyDeviceToDevice, stream);
  hipMemcpyAsync(c_out, cbuf[T_STEPS & 1], HC * sizeof(float),
                 hipMemcpyDeviceToDevice, stream);
}
